// AttentivePoolingNetwork_52604759441513
// MI455X (gfx1250) — matrix operand order repaired and checked by reading, not compiled, not run on hardware
//
#include <hip/hip_runtime.h>
#include <hip/hip_bf16.h>

// ---------------------------------------------------------------------------
// AttentivePoolingNetwork on MI455X (gfx1250), bf16 WMMA path.
//   B=512, QL=128, AL=512, V=50000, E=300 (pad 320), F=400 (pad 416), K=3
// Matmuls: v_wmma_f32_16x16x32_bf16 (f32 accumulate).
// Shared operands staged into LDS with global_load_async_to_lds_b128
// (ASYNCcnt) and read back as WMMA fragments via ds_load_b128.
// ---------------------------------------------------------------------------

typedef __bf16 bf16;
typedef __attribute__((ext_vector_type(16))) __bf16 v16bf;
typedef __attribute__((ext_vector_type(8)))  __bf16 v8bf;
typedef __attribute__((ext_vector_type(8)))  float  v8f;

#define Bc   512
#define QLc  128
#define ALc  512
#define Ec   300
#define Epc  320   // E padded to multiple of 32 (WMMA K)
#define Fc   400
#define Fpc  416   // F padded to multiple of 32 (WMMA K) and 16 (tiles)
#define FT   25    // F / 16 tiles
#define GT   26    // Fp / 16 tiles

__device__ __forceinline__ v16bf frag_runs(const bf16* p0, const bf16* p1) {
    v8bf a = *(const v8bf*)p0;
    v8bf b = *(const v8bf*)p1;
    return __builtin_shufflevector(a, b, 0,1,2,3,4,5,6,7,8,9,10,11,12,13,14,15);
}

// Async copy 16 bytes global -> LDS (tracked by ASYNCcnt, no VGPR transit).
__device__ __forceinline__ void async_load_b128(unsigned lds_off, const bf16* gaddr) {
    asm volatile("global_load_async_to_lds_b128 %0, %1, off"
                 :: "v"(lds_off), "v"(gaddr)
                 : "memory");
}

__device__ __forceinline__ void wait_async0() {
    asm volatile("s_wait_asynccnt 0" ::: "memory");
}

__device__ __forceinline__ void atomicMaxF(float* addr, float v) {
    unsigned int* ua = (unsigned int*)addr;
    unsigned int old = *ua;
    while (true) {
        float f = __uint_as_float(old);
        if (f >= v) break;
        unsigned int assumed = old;
        old = atomicCAS(ua, assumed, __float_as_uint(v));
        if (old == assumed) break;
    }
}

// ---------------------------------------------------------------------------
// Prep kernels
// ---------------------------------------------------------------------------

// conv_w [F,E,K] f32 -> wbf [K][F][Ep] bf16 (e >= E zero)
__global__ void prep_w_kernel(const float* __restrict__ w, bf16* __restrict__ wbf) {
    int i = blockIdx.x * blockDim.x + threadIdx.x;
    if (i >= 3 * Fc * Epc) return;
    int e = i % Epc;
    int f = (i / Epc) % Fc;
    int k = i / (Epc * Fc);
    float v = (e < Ec) ? w[(f * Ec + e) * 3 + k] : 0.f;
    wbf[i] = (bf16)v;
}

// U [F,F] f32 -> Ut [Fp][Fp] bf16, transposed: Ut[g*Fp+f] = U[f*F+g]; pads zero
__global__ void prep_u_kernel(const float* __restrict__ U, bf16* __restrict__ Ut) {
    int i = blockIdx.x * blockDim.x + threadIdx.x;
    if (i >= Fpc * Fpc) return;
    int f = i % Fpc;
    int g = i / Fpc;
    float v = (f < Fc && g < Fc) ? U[f * Fc + g] : 0.f;
    Ut[i] = (bf16)v;
}

// tokens [P] int64, emb [V+1, E] -> out [P][Ep] bf16 (e >= E zero)
__global__ void gather_embed_kernel(const long long* __restrict__ tok,
                                    const float* __restrict__ emb,
                                    bf16* __restrict__ out, long long npos) {
    size_t i = (size_t)blockIdx.x * blockDim.x + threadIdx.x;
    if (i >= (size_t)npos * Epc) return;
    size_t pos = i / Epc;
    int e = (int)(i % Epc);
    long long t = tok[pos];
    float v = (e < Ec) ? emb[(size_t)t * Ec + e] : 0.f;
    out[i] = (bf16)v;
}

// zero the f in [400,416) pad columns of a [rows][Fp] bf16 buffer
__global__ void pad_zero_kernel(bf16* __restrict__ p, int rows) {
    int i = blockIdx.x * blockDim.x + threadIdx.x;
    if (i >= rows * 16) return;
    int r = i / 16, c = i % 16;
    p[(size_t)r * Fpc + Fc + c] = (bf16)0.f;
}

__global__ void fill_f32_kernel(float* __restrict__ p, float v, int n) {
    int i = blockIdx.x * blockDim.x + threadIdx.x;
    if (i < n) p[i] = v;
}

// ---------------------------------------------------------------------------
// Conv1d (k=3, pad=1) as 3 shifted GEMMs.
// Block = 8 waves; block owns one f-tile (W rows async-staged in LDS once),
// wave w owns l-tile (blockIdx.y/FT)*8 + w.
//   X   : [B][L][Ep] bf16 ; Wb : [3][F][Ep] bf16 ; Out : [B][L][Fp] bf16
// ---------------------------------------------------------------------------
__global__ __launch_bounds__(256) void conv_gemm_kernel(
        const bf16* __restrict__ X, const bf16* __restrict__ Wb,
        const float* __restrict__ bias, bf16* __restrict__ Out, int L) {
    __shared__ __align__(16) bf16 sW[3 * 16 * Epc];   // 30 KB: 3 taps x 16 f-rows

    const int b      = blockIdx.x;
    const int ftile  = blockIdx.y % FT;
    const int lgroup = blockIdx.y / FT;
    const int wave   = threadIdx.x >> 5;
    const int ltile  = lgroup * 8 + wave;
    const int lane   = threadIdx.x & 31;
    const int m      = lane & 15;                 // A-frag row (f)
    const int n      = lane & 15;                 // B-frag / C col (l)
    const int kb     = (lane < 16) ? 0 : 8;       // A-frag K sub-offset
    const int kb2    = (lane < 16) ? 0 : 16;      // unused; B-frag uses kb as well
    const int col    = ltile * 16 + n;            // output position l

    // Async-stage W tile: 3 contiguous chunks of 16*Epc bf16 (10240 B each).
    for (int idx = threadIdx.x; idx < 3 * 16 * Epc / 8; idx += 256) {
        const int k = idx / (16 * Epc / 8);
        const int rem = idx % (16 * Epc / 8);
        const bf16* gsrc = Wb + ((size_t)k * Fc + ftile * 16) * Epc + rem * 8;
        async_load_b128((unsigned)(size_t)&sW[k * 16 * Epc + rem * 8], gsrc);
    }
    wait_async0();
    __syncthreads();

    v8f acc = {};
    #pragma unroll
    for (int k = 0; k < 3; k++) {
        const int row = col + k - 1;
        const bool ok = (row >= 0) && (row < L);
        const bf16* xr = X + ((size_t)b * L + (ok ? row : 0)) * Epc;
        const bf16* wk = &sW[(k * 16 + m) * Epc];
        #pragma unroll
        for (int ec = 0; ec < Epc; ec += 32) {
            v16bf fa = frag_runs(wk + ec + kb, wk + ec + 16 + kb);   // ds_load
            v16bf fb = {};
            if (ok) fb = frag_runs(xr + ec + kb, xr + ec + 16 + kb);
            acc = __builtin_amdgcn_wmma_f32_16x16x32_bf16(
                false, fa, false, fb, (short)0, acc, false, false);
        }
    }

    // C layout: lane n fixed; rows f = mb + r -> contiguous in [l][f] layout.
    const int mb = (lane < 16) ? 0 : 8;
    const float* bp = bias + ftile * 16 + mb;
    v8bf o;
    #pragma unroll
    for (int r = 0; r < 8; r++) o[r] = (bf16)(acc[r] + bp[r]);
    *(v8bf*)(Out + ((size_t)b * L + col) * Fpc + ftile * 16 + mb) = o;
}

// ---------------------------------------------------------------------------
// M1[b] = Q[b]^T (QL x F) @ U (F x F)  ->  [B][QL][Fp] bf16
// Block owns g-tile (Ut rows async-staged in LDS), wave owns q-tile.
// gtile runs 0..25 so pad columns of M1 are computed (exact zeros).
// ---------------------------------------------------------------------------
__global__ __launch_bounds__(256) void gemm_qtu_kernel(
        const bf16* __restrict__ Qbf, const bf16* __restrict__ Ut,
        bf16* __restrict__ M1) {
    __shared__ __align__(16) bf16 sB[16 * Fpc];   // 13 KB: 16 g-rows of Ut

    const int b     = blockIdx.x;
    const int gtile = blockIdx.y;            // 0..25
    const int qtile = threadIdx.x >> 5;      // 0..7
    const int lane  = threadIdx.x & 31;
    const int mn    = lane & 15;
    const int kb    = (lane < 16) ? 0 : 8;
    const int kb2   = (lane < 16) ? 0 : 16;

    const bf16* gB = Ut + (size_t)(gtile * 16) * Fpc;   // contiguous 16 rows
    for (int idx = threadIdx.x; idx < 16 * Fpc / 8; idx += 256)
        async_load_b128((unsigned)(size_t)&sB[idx * 8], gB + idx * 8);
    wait_async0();
    __syncthreads();

    const bf16* Arow = Qbf + ((size_t)b * QLc + qtile * 16 + mn) * Fpc;
    const bf16* Brow = &sB[mn * Fpc];

    v8f acc = {};
    #pragma unroll
    for (int kc = 0; kc < Fpc; kc += 32) {
        v16bf fa = frag_runs(Arow + kc + kb, Arow + kc + 16 + kb);
        v16bf fb = frag_runs(Brow + kc + kb, Brow + kc + 16 + kb); // ds_load
        acc = __builtin_amdgcn_wmma_f32_16x16x32_bf16(
            false, fa, false, fb, (short)0, acc, false, false);
    }

    const int mb = (lane < 16) ? 0 : 8;
    bf16* op = M1 + ((size_t)b * QLc + qtile * 16 + mb) * Fpc + gtile * 16 + mn;
    #pragma unroll
    for (int r = 0; r < 8; r++) op[(size_t)r * Fpc] = (bf16)acc[r];
}

// ---------------------------------------------------------------------------
// G[b] = tanh(M1[b] (QL x Fp) @ A[b]^T (Fp x AL)) — fused: never stored.
// Block owns a-tile (A rows async-staged in LDS), wave owns q-tile.
// Row/col maxima published with float atomic-max (CAS loop).
// ---------------------------------------------------------------------------
__global__ __launch_bounds__(256) void gemm_g_fused_kernel(
        const bf16* __restrict__ M1, const bf16* __restrict__ Abf,
        float* __restrict__ rowmax, float* __restrict__ colmax) {
    __shared__ __align__(16) bf16 sB[16 * Fpc];   // 13 KB: 16 a-rows of A[b]

    const int b     = blockIdx.x;
    const int atile = blockIdx.y;            // 0..31
    const int qtile = threadIdx.x >> 5;      // 0..7
    const int lane  = threadIdx.x & 31;
    const int mn    = lane & 15;
    const int kb    = (lane < 16) ? 0 : 8;
    const int kb2   = (lane < 16) ? 0 : 16;

    const bf16* gB = Abf + ((size_t)b * ALc + atile * 16) * Fpc; // contiguous
    for (int idx = threadIdx.x; idx < 16 * Fpc / 8; idx += 256)
        async_load_b128((unsigned)(size_t)&sB[idx * 8], gB + idx * 8);
    wait_async0();
    __syncthreads();

    const bf16* Arow = M1 + ((size_t)b * QLc + qtile * 16 + mn) * Fpc;
    const bf16* Brow = &sB[mn * Fpc];

    v8f acc = {};
    #pragma unroll
    for (int kc = 0; kc < Fpc; kc += 32) {
        v16bf fa = frag_runs(Arow + kc + kb, Arow + kc + 16 + kb);
        v16bf fb = frag_runs(Brow + kc + kb, Brow + kc + 16 + kb); // ds_load
        acc = __builtin_amdgcn_wmma_f32_16x16x32_bf16(
            false, fa, false, fb, (short)0, acc, false, false);
    }

    float t[8];
    #pragma unroll
    for (int r = 0; r < 8; r++) t[r] = tanhf(acc[r]);

    const int mb = (lane < 16) ? 0 : 8;

    // Row max (over a): reduce each r across the 16 lanes of this half-wave.
    #pragma unroll
    for (int r = 0; r < 8; r++) {
        float v = t[r];
        v = fmaxf(v, __shfl_xor(v, 1, 32));
        v = fmaxf(v, __shfl_xor(v, 2, 32));
        v = fmaxf(v, __shfl_xor(v, 4, 32));
        v = fmaxf(v, __shfl_xor(v, 8, 32));
        if (mn == 0)
            atomicMaxF(rowmax + (size_t)b * QLc + qtile * 16 + mb + r, v);
    }

    // Col max (over q): max over this lane's 8 rows, combine halves.
    float cm = t[0];
    #pragma unroll
    for (int r = 1; r < 8; r++) cm = fmaxf(cm, t[r]);
    cm = fmaxf(cm, __shfl_xor(cm, 16, 32));
    if (lane < 16)
        atomicMaxF(colmax + (size_t)b * ALc + atile * 16 + mn, cm);
}

// ---------------------------------------------------------------------------
// Softmax over each row of src [B][n]  (blockDim.x == n, n <= 512)
// ---------------------------------------------------------------------------
__global__ void softmax_kernel(const float* __restrict__ src,
                               float* __restrict__ dst, int n) {
    __shared__ float sm[512];
    const int b = blockIdx.x, t = threadIdx.x;
    float v = src[(size_t)b * n + t];
    sm[t] = v; __syncthreads();
    for (int s = n >> 1; s > 0; s >>= 1) {
        if (t < s) sm[t] = fmaxf(sm[t], sm[t + s]);
        __syncthreads();
    }
    const float mx = sm[0]; __syncthreads();
    const float e = __expf(v - mx);
    sm[t] = e; __syncthreads();
    for (int s = n >> 1; s > 0; s >>= 1) {
        if (t < s) sm[t] += sm[t + s];
        __syncthreads();
    }
    dst[(size_t)b * n + t] = e / sm[0];
}

// rOut[b][f] = sum_l X[b][l][f] * w[b][l]   (X bf16 [B][Lw][Fp])
__global__ void weighted_pool_kernel(const bf16* __restrict__ X,
                                     const float* __restrict__ w,
                                     float* __restrict__ rOut, int Lw) {
    const int b = blockIdx.x;
    const int f = blockIdx.y * 128 + threadIdx.x;
    if (f >= Fc) return;
    const bf16* xb = X + (size_t)b * Lw * Fpc + f;
    const float* wb = w + (size_t)b * Lw;
    float acc = 0.f;
    for (int l = 0; l < Lw; l++)
        acc += (float)xb[(size_t)l * Fpc] * wb[l];
    rOut[(size_t)b * Fc + f] = acc;
}

// out[b] = dot(rQ,rA) / (max(|rQ|,eps) * max(|rA|,eps))
__global__ void cosine_kernel(const float* __restrict__ rQ,
                              const float* __restrict__ rA,
                              float* __restrict__ out) {
    __shared__ float sd[128], sq[128], sa[128];
    const int b = blockIdx.x, t = threadIdx.x;
    float d = 0.f, q = 0.f, a = 0.f;
    for (int f = t; f < Fc; f += 128) {
        float x = rQ[(size_t)b * Fc + f];
        float y = rA[(size_t)b * Fc + f];
        d += x * y; q += x * x; a += y * y;
    }
    sd[t] = d; sq[t] = q; sa[t] = a; __syncthreads();
    for (int s = 64; s > 0; s >>= 1) {
        if (t < s) { sd[t] += sd[t+s]; sq[t] += sq[t+s]; sa[t] += sa[t+s]; }
        __syncthreads();
    }
    if (t == 0) {
        float nq = fmaxf(sqrtf(sq[0]), 1e-8f);
        float na = fmaxf(sqrtf(sa[0]), 1e-8f);
        out[b] = sd[0] / (nq * na);
    }
}

// ---------------------------------------------------------------------------
// Launch
// ---------------------------------------------------------------------------
extern "C" void kernel_launch(void* const* d_in, const int* in_sizes, int n_in,
                              void* d_out, int out_size, void* d_ws, size_t ws_size,
                              hipStream_t stream) {
    const long long* question = (const long long*)d_in[0]; // [B, QL] int64
    const long long* answer   = (const long long*)d_in[1]; // [B, AL] int64
    const float* emb   = (const float*)d_in[2];            // [V+1, E]
    const float* convw = (const float*)d_in[3];            // [F, E, 3]
    const float* convb = (const float*)d_in[4];            // [F]
    const float* U     = (const float*)d_in[5];            // [F, F]
    float* out = (float*)d_out;                            // [B]

    char* base = (char*)d_ws;
    size_t off = 0;
    auto carve = [&](size_t bytes) -> void* {
        void* p = base + off;
        off = (off + bytes + 255) & ~(size_t)255;
        return p;
    };

    bf16*  qx     = (bf16*) carve((size_t)Bc * QLc * Epc * 2);  //  42 MB
    bf16*  ax     = (bf16*) carve((size_t)Bc * ALc * Epc * 2);  // 168 MB
    bf16*  wbf    = (bf16*) carve((size_t)3 * Fc * Epc * 2);
    bf16*  Ut     = (bf16*) carve((size_t)Fpc * Fpc * 2);
    bf16*  Qbf    = (bf16*) carve((size_t)Bc * QLc * Fpc * 2);  //  55 MB
    bf16*  Abf    = (bf16*) carve((size_t)Bc * ALc * Fpc * 2);  // 218 MB
    bf16*  M1     = (bf16*) carve((size_t)Bc * QLc * Fpc * 2);  //  55 MB
    float* rowmax = (float*)carve((size_t)Bc * QLc * 4);
    float* colmax = (float*)carve((size_t)Bc * ALc * 4);
    float* roQ    = (float*)carve((size_t)Bc * QLc * 4);
    float* roA    = (float*)carve((size_t)Bc * ALc * 4);
    float* rQ     = (float*)carve((size_t)Bc * Fc * 4);
    float* rA     = (float*)carve((size_t)Bc * Fc * 4);

    // --- prep ---
    {
        int n = 3 * Fc * Epc;
        prep_w_kernel<<<(n + 255) / 256, 256, 0, stream>>>(convw, wbf);
    }
    {
        int n = Fpc * Fpc;
        prep_u_kernel<<<(n + 255) / 256, 256, 0, stream>>>(U, Ut);
    }
    {
        long long nposQ = (long long)Bc * QLc;
        size_t tq = (size_t)nposQ * Epc;
        gather_embed_kernel<<<(unsigned)((tq + 255) / 256), 256, 0, stream>>>(
            question, emb, qx, nposQ);
        long long nposA = (long long)Bc * ALc;
        size_t ta = (size_t)nposA * Epc;
        gather_embed_kernel<<<(unsigned)((ta + 255) / 256), 256, 0, stream>>>(
            answer, emb, ax, nposA);
    }
    {
        int rq = Bc * QLc, ra = Bc * ALc;
        pad_zero_kernel<<<(rq * 16 + 255) / 256, 256, 0, stream>>>(Qbf, rq);
        pad_zero_kernel<<<(ra * 16 + 255) / 256, 256, 0, stream>>>(Abf, ra);
        fill_f32_kernel<<<(rq + 255) / 256, 256, 0, stream>>>(rowmax, -2.0f, rq);
        fill_f32_kernel<<<(ra + 255) / 256, 256, 0, stream>>>(colmax, -2.0f, ra);
    }

    // --- conv as GEMM (WMMA): grid.y = FT * (L/128) l-groups ---
    conv_gemm_kernel<<<dim3(Bc, FT * (QLc / 128)), 256, 0, stream>>>(
        qx, wbf, convb, Qbf, QLc);
    conv_gemm_kernel<<<dim3(Bc, FT * (ALc / 128)), 256, 0, stream>>>(
        ax, wbf, convb, Abf, ALc);

    // --- M1 = Q^T U (WMMA) ---
    gemm_qtu_kernel<<<dim3(Bc, GT), 256, 0, stream>>>(Qbf, Ut, M1);

    // --- G = tanh(M1 A^T), fused row/col max (WMMA) ---
    gemm_g_fused_kernel<<<dim3(Bc, ALc / 16), 256, 0, stream>>>(
        M1, Abf, rowmax, colmax);

    // --- softmax ---
    softmax_kernel<<<Bc, QLc, 0, stream>>>(rowmax, roQ, QLc);
    softmax_kernel<<<Bc, ALc, 0, stream>>>(colmax, roA, ALc);

    // --- weighted pools ---
    weighted_pool_kernel<<<dim3(Bc, (Fc + 127) / 128), 128, 0, stream>>>(
        Qbf, roQ, rQ, QLc);
    weighted_pool_kernel<<<dim3(Bc, (Fc + 127) / 128), 128, 0, stream>>>(
        Abf, roA, rA, ALc);

    // --- cosine similarity ---
    cosine_kernel<<<Bc, 128, 0, stream>>>(rQ, rA, out);
}
